// TransitionUp_54786602828255
// MI455X (gfx1250) — matrix operand order repaired and checked by reading, not compiled, not run on hardware
//
#include <hip/hip_runtime.h>
#include <hip/hip_bf16.h>

// ---------------------------------------------------------------------------
// Types for CDNA5 WMMA (wave32) + TDM descriptor groups
// ---------------------------------------------------------------------------
typedef __bf16       v16bf __attribute__((ext_vector_type(16)));
typedef float        v8f   __attribute__((ext_vector_type(8)));
typedef unsigned int u32x4 __attribute__((ext_vector_type(4)));
typedef int          i32x4 __attribute__((ext_vector_type(4)));
typedef int          i32x8 __attribute__((ext_vector_type(8)));

union BFrag {
    v16bf        v;
    unsigned int u32[8];
};

__device__ __forceinline__ unsigned short f2bf(float f) {
    // round-to-nearest-even f32 -> bf16 (bit pattern)
    unsigned int u = __float_as_uint(f);
    u += 0x7FFFu + ((u >> 16) & 1u);
    return (unsigned short)(u >> 16);
}

#define NOUT 256   // output channels (Cout) -- fixed by problem

#if __has_builtin(__builtin_amdgcn_tensor_load_to_lds)
#define USE_TDM 1
#else
#define USE_TDM 0
#endif

#if USE_TDM
// ---------------------------------------------------------------------------
// Issue one TDM 2D tile load: tile_d1 rows x tile_d0 elements of 2-byte data,
// row stride = stride0 elements, packed contiguously into LDS at lds_off.
// Descriptor bitfields per CDNA5 ISA §8 (D# group 0 / group 1).
// ---------------------------------------------------------------------------
__device__ __forceinline__ void tdm_load_2d_bf16(unsigned lds_off,
                                                 const unsigned short* gptr,
                                                 unsigned tensor_d0, unsigned tensor_d1,
                                                 unsigned tile_d0,   unsigned tile_d1,
                                                 unsigned stride0)
{
    unsigned long long ga = (unsigned long long)gptr;

    u32x4 g0;
    g0[0] = 1u;                                   // count=1, user mode, no gather
    g0[1] = lds_off;                              // LDS byte address
    g0[2] = (unsigned)ga;                         // global_addr[31:0]
    g0[3] = (unsigned)(ga >> 32) | (2u << 30);    // global_addr[56:32] | type=2

    i32x8 g1;
    g1[0] = (int)(1u << 16);                                      // data_size=1 (2B); wg_mask=0
    g1[1] = (int)((tensor_d0 & 0xFFFFu) << 16);                   // tensor_dim0[15:0]
    g1[2] = (int)((tensor_d0 >> 16) | ((tensor_d1 & 0xFFFFu) << 16)); // dim0[31:16] | dim1[15:0]
    g1[3] = (int)((tensor_d1 >> 16) | (tile_d0 << 16));           // dim1[31:16] | tile_dim0
    g1[4] = (int)(tile_d1 & 0xFFFFu);                             // tile_dim1 | tile_dim2=0
    g1[5] = (int)stride0;                                         // tensor_dim0_stride[31:0]
    g1[6] = 0;                                                    // stride0[47:32] | stride1[15:0]
    g1[7] = 0;

    i32x4 z4 = {0, 0, 0, 0};
#if __has_include(<hip/amd_detail/amd_gfx1250_TDM.h>)
    i32x8 z8 = {0, 0, 0, 0, 0, 0, 0, 0};
    __builtin_amdgcn_tensor_load_to_lds(g0, g1, z4, z4, z8, 0);   // 6-arg toolchain
#else
    __builtin_amdgcn_tensor_load_to_lds(g0, g1, z4, z4, 0);       // 5-arg ROCm 7.2
#endif
}
#endif // USE_TDM

// ---------------------------------------------------------------------------
// Pre-pass: f32 -> bf16 conversion (vectorized), and W transpose+convert.
// Everything stays L2-resident afterwards (total footprint << 192 MB).
// ---------------------------------------------------------------------------
__global__ __launch_bounds__(256)
void convert_bf16_vec4(const float* __restrict__ in, unsigned short* __restrict__ out, int n4)
{
    int i = blockIdx.x * 256 + threadIdx.x;
    if (i < n4) {
        float4 f = ((const float4*)in)[i];
        unsigned int lo = (unsigned int)f2bf(f.x) | ((unsigned int)f2bf(f.y) << 16);
        unsigned int hi = (unsigned int)f2bf(f.z) | ((unsigned int)f2bf(f.w) << 16);
        ((uint2*)out)[i] = make_uint2(lo, hi);
    }
}

__global__ __launch_bounds__(256)
void transpose_w_bf16(const float* __restrict__ W, unsigned short* __restrict__ Wt, int K)
{
    // W: [K][256] row-major  ->  Wt: [256][K]
    int idx = blockIdx.x * 256 + threadIdx.x;   // K*256 total
    int k = idx >> 8, n = idx & 255;
    Wt[(size_t)n * K + k] = f2bf(W[idx]);
}

// ---------------------------------------------------------------------------
// GEMM: Y[M][256] = A[M][K](bf16) @ Wt[256][K](bf16)^T + bias, fused BN stats.
// Block: 128 threads = 4 waves, tile 64(M) x 64(N), K-step 32.
// Tiles staged by the Tensor Data Mover into double-buffered LDS; wave 0
// issues TDM for step i+1 while all waves run WMMA on step i.
// ---------------------------------------------------------------------------
__global__ __launch_bounds__(128)
void gemm_tdm_bias_stats(const unsigned short* __restrict__ Abf,
                         const unsigned short* __restrict__ Wt,
                         const float* __restrict__ bias, float* __restrict__ Y,
                         float* __restrict__ stats, int M, int K)
{
    __shared__ unsigned short As[2][64 * 32];   // [m][k] bf16
    __shared__ unsigned short Bs[2][64 * 32];   // [n][k] bf16

    const int tid  = threadIdx.x;
    const int lane = tid & 31;
    const int wv   = tid >> 5;
    const int m0   = blockIdx.x * 64;
    const int n0   = blockIdx.y * 64;
    const int l15  = lane & 15;
    const int g    = lane >> 4;
    const int nsteps = K >> 5;

#if USE_TDM
    const unsigned asOff0 = (unsigned)(unsigned long long)(void*)&As[0][0];
    const unsigned asOff1 = (unsigned)(unsigned long long)(void*)&As[1][0];
    const unsigned bsOff0 = (unsigned)(unsigned long long)(void*)&Bs[0][0];
    const unsigned bsOff1 = (unsigned)(unsigned long long)(void*)&Bs[1][0];

    if (tid < 32) {   // wave 0 drives the TDM pipeline
        tdm_load_2d_bf16(asOff0, Abf + (size_t)m0 * K, K, M, 32, 64, K);
        tdm_load_2d_bf16(bsOff0, Wt  + (size_t)n0 * K, K, NOUT, 32, 64, K);
    }
#endif

    v8f acc[4] = {};

    for (int i = 0; i < nsteps; ++i) {
        const int cur = i & 1;
        __syncthreads();   // all waves done reading buffer cur^1 (prev step)

#if USE_TDM
        if (tid < 32) {
            if (i + 1 < nsteps) {
                const int k1 = (i + 1) << 5;
                unsigned aoff = cur ? asOff0 : asOff1;   // next buffer
                unsigned boff = cur ? bsOff0 : bsOff1;
                tdm_load_2d_bf16(aoff, Abf + (size_t)m0 * K + k1, K, M, 32, 64, K);
                tdm_load_2d_bf16(boff, Wt  + (size_t)n0 * K + k1, K, NOUT, 32, 64, K);
                __builtin_amdgcn_s_wait_tensorcnt(2);    // current buffer complete
            } else {
                __builtin_amdgcn_s_wait_tensorcnt(0);
            }
        }
#else
        // Fallback: cooperative staging of pre-converted bf16 (8 ushorts/chunk)
        {
            const int k0 = i << 5;
            #pragma unroll
            for (int c = 0; c < 2; ++c) {
                int ch = tid + c * 128;                 // 256 chunks of 8
                int r = ch >> 2, ko = (ch & 3) * 8;
                *(uint4*)&As[cur][r * 32 + ko] = *(const uint4*)&Abf[(size_t)(m0 + r) * K + k0 + ko];
                *(uint4*)&Bs[cur][r * 32 + ko] = *(const uint4*)&Wt [(size_t)(n0 + r) * K + k0 + ko];
            }
        }
#endif
        __syncthreads();   // buffer cur visible to all waves

        // A fragment: lane half g covers K {0..7,16..23} / {8..15,24..31}
        BFrag a;
        const int arow = wv * 16 + l15;
        #pragma unroll
        for (int j = 0; j < 16; j += 2) {
            int kA = (j < 8) ? (j + 8 * g) : (j + 8 + 8 * g);
            a.u32[j >> 1] = *(const unsigned int*)(&As[cur][arow * 32 + kA]);
        }
        // B fragments: lane half g covers K {0..7,16..23} / {8..15,24..31}
        #pragma unroll
        for (int s = 0; s < 4; ++s) {
            BFrag b;
            const int ncol = s * 16 + l15;
            #pragma unroll
            for (int j = 0; j < 16; j += 2) {
                int kB = (j < 8) ? (j + 8 * g) : (j + 8 + 8 * g);
                b.u32[j >> 1] = *(const unsigned int*)(&Bs[cur][ncol * 32 + kB]);
            }
            acc[s] = __builtin_amdgcn_wmma_f32_16x16x32_bf16(
                false, a.v, false, b.v, (short)0, acc[s], false, false);
        }
    }

    // Epilogue: bias, store Y, fused BN statistics (sum / sum^2 per column)
    #pragma unroll
    for (int s = 0; s < 4; ++s) {
        const int col = n0 + s * 16 + l15;
        const float bv = bias[col];
        float csum = 0.f, csq = 0.f;
        #pragma unroll
        for (int r = 0; r < 8; ++r) {
            const int row = m0 + wv * 16 + r + 8 * g;   // D: m = r + 8*half
            float v = acc[s][r] + bv;
            Y[(size_t)row * NOUT + col] = v;
            csum += v;
            csq  += v * v;
        }
        csum += __shfl_xor(csum, 16, 32);
        csq  += __shfl_xor(csq, 16, 32);
        if (lane < 16) {
            atomicAdd(&stats[col],        csum);
            atomicAdd(&stats[NOUT + col], csq);
        }
    }
}

// ---------------------------------------------------------------------------
// Fold BN statistics into per-channel scale/shift.
// ---------------------------------------------------------------------------
__global__ __launch_bounds__(NOUT)
void finalize_stats(const float* __restrict__ stats,
                    const float* __restrict__ gamma,
                    const float* __restrict__ beta,
                    float* __restrict__ scale, float* __restrict__ shift,
                    float invM)
{
    const int c = threadIdx.x;
    float mean = stats[c] * invM;
    float var  = stats[NOUT + c] * invM - mean * mean;   // biased variance
    float is   = rsqrtf(var + 1e-5f);
    float sc   = gamma[c] * is;
    scale[c] = sc;
    shift[c] = beta[c] - mean * sc;
}

// ---------------------------------------------------------------------------
// Brute-force 3-NN: one thread per fine point; coarse points staged in LDS as
// float4(x,y,z,|p|^2) chunks of 512. Rank on key = |p2|^2 - 2*dot.
// ---------------------------------------------------------------------------
__global__ __launch_bounds__(256)
void knn3_kernel(const float* __restrict__ p1, const float* __restrict__ p2,
                 int N1, int N2, int* __restrict__ idx, float* __restrict__ wgt)
{
    __shared__ float4 pc[512];
    const int tid = threadIdx.x;
    const int bpb = N1 >> 8;
    const int b   = blockIdx.x / bpb;
    const int n   = (blockIdx.x % bpb) * 256 + tid;

    const float* q = &p1[((size_t)b * N1 + n) * 3];
    const float qx = q[0], qy = q[1], qz = q[2];
    const float qsq = qx * qx + qy * qy + qz * qz;

    float k0 = 3.4e38f, k1 = 3.4e38f, k2 = 3.4e38f;
    int   i0 = 0, i1 = 0, i2 = 0;

    for (int c0 = 0; c0 < N2; c0 += 512) {
        __syncthreads();
        #pragma unroll
        for (int i = 0; i < 2; ++i) {
            int pi = tid + i * 256;
            const float* pp = &p2[((size_t)b * N2 + c0 + pi) * 3];
            float x = pp[0], y = pp[1], z = pp[2];
            pc[pi] = make_float4(x, y, z, x * x + y * y + z * z);
        }
        __syncthreads();

        #pragma unroll 4
        for (int j = 0; j < 512; ++j) {
            float4 t  = pc[j];
            float dot = qx * t.x + qy * t.y + qz * t.z;
            float key = __builtin_fmaf(-2.f, dot, t.w);   // d2 - qsq
            int   id  = c0 + j;
            if (key < k2) {
                if (key < k1) {
                    k2 = k1; i2 = i1;
                    if (key < k0) { k1 = k0; i1 = i0; k0 = key; i0 = id; }
                    else          { k1 = key; i1 = id; }
                } else { k2 = key; i2 = id; }
            }
        }
    }

    float d0 = sqrtf(fmaxf(k0 + qsq, 0.f));
    float d1 = sqrtf(fmaxf(k1 + qsq, 0.f));
    float d2 = sqrtf(fmaxf(k2 + qsq, 0.f));
    float r0 = 1.f / (d0 + 1e-8f);
    float r1 = 1.f / (d1 + 1e-8f);
    float r2 = 1.f / (d2 + 1e-8f);
    float inv = 1.f / (r0 + r1 + r2);

    const size_t base = ((size_t)b * N1 + n) * 3;
    idx[base + 0] = b * N2 + i0;
    idx[base + 1] = b * N2 + i1;
    idx[base + 2] = b * N2 + i2;
    wgt[base + 0] = r0 * inv;
    wgt[base + 1] = r1 * inv;
    wgt[base + 2] = r2 * inv;
}

// ---------------------------------------------------------------------------
// Gather + inverse-distance interpolate + skip path (BN+ReLU on the fly).
// ---------------------------------------------------------------------------
__global__ __launch_bounds__(NOUT)
void interp_skip_kernel(const float* __restrict__ y1, const float* __restrict__ y2,
                        const int* __restrict__ idx, const float* __restrict__ wgt,
                        const float* __restrict__ sc1, const float* __restrict__ sh1,
                        const float* __restrict__ sc2, const float* __restrict__ sh2,
                        float* __restrict__ out)
{
    const int    pt = blockIdx.x;
    const int    c  = threadIdx.x;
    const size_t b3 = (size_t)pt * 3;

    const int   j0 = idx[b3 + 0], j1 = idx[b3 + 1], j2 = idx[b3 + 2];
    const float w0 = wgt[b3 + 0], w1 = wgt[b3 + 1], w2 = wgt[b3 + 2];

    const float s2c = sc2[c], h2c = sh2[c];
    float f0 = fmaxf(y2[(size_t)j0 * NOUT + c] * s2c + h2c, 0.f);
    float f1 = fmaxf(y2[(size_t)j1 * NOUT + c] * s2c + h2c, 0.f);
    float f2 = fmaxf(y2[(size_t)j2 * NOUT + c] * s2c + h2c, 0.f);
    float sk = fmaxf(y1[(size_t)pt * NOUT + c] * sc1[c] + sh1[c], 0.f);

    out[(size_t)pt * NOUT + c] = w0 * f0 + w1 * f1 + w2 * f2 + sk;
}

// ---------------------------------------------------------------------------
// Host-side launcher
// ---------------------------------------------------------------------------
extern "C" void kernel_launch(void* const* d_in, const int* in_sizes, int n_in,
                              void* d_out, int out_size, void* d_ws, size_t ws_size,
                              hipStream_t stream)
{
    const float* p1  = (const float*)d_in[0];
    const float* x1  = (const float*)d_in[1];
    const float* p2  = (const float*)d_in[2];
    const float* x2  = (const float*)d_in[3];
    const float* W1  = (const float*)d_in[4];
    const float* b1  = (const float*)d_in[5];
    const float* g1  = (const float*)d_in[6];
    const float* be1 = (const float*)d_in[7];
    const float* W2  = (const float*)d_in[8];
    const float* b2  = (const float*)d_in[9];
    const float* g2  = (const float*)d_in[10];
    const float* be2 = (const float*)d_in[11];

    const int B    = 2;
    const int Cout = in_sizes[5];              // 256
    const int Cin  = in_sizes[8] / Cout;       // 512
    const int N1   = in_sizes[0] / (3 * B);    // 16384
    const int N2   = in_sizes[2] / (3 * B);    // 4096
    const int M1   = B * N1;                   // 32768
    const int M2   = B * N2;                   // 8192

    // ---- workspace layout ----
    float* ws  = (float*)d_ws;
    float* y1  = ws;                               // M1 * 256 f32
    float* y2  = y1 + (size_t)M1 * NOUT;           // M2 * 256 f32
    float* st1 = y2 + (size_t)M2 * NOUT;           // 512 (sum, sumsq)
    float* st2 = st1 + 2 * NOUT;                   // 512
    float* sc1 = st2 + 2 * NOUT;
    float* sh1 = sc1 + NOUT;
    float* sc2 = sh1 + NOUT;
    float* sh2 = sc2 + NOUT;
    int*   nni = (int*)(sh2 + NOUT);               // B*N1*3
    float* nnw = (float*)(nni + (size_t)B * N1 * 3);
    unsigned short* a1b = (unsigned short*)(nnw + (size_t)B * N1 * 3); // M1*Cout bf16
    unsigned short* a2b = a1b + (size_t)M1 * Cout;                     // M2*Cin  bf16
    unsigned short* w1t = a2b + (size_t)M2 * Cin;                      // 256*Cout bf16 (transposed)
    unsigned short* w2t = w1t + (size_t)NOUT * Cout;                   // 256*Cin  bf16 (transposed)

    hipMemsetAsync(st1, 0, 4 * NOUT * sizeof(float), stream);

    // ---- pre-pass: bf16 conversions (+ weight transpose for TDM-friendly B tiles)
    {
        int n4 = (M1 * Cout) / 4;
        convert_bf16_vec4<<<(n4 + 255) / 256, 256, 0, stream>>>(x1, a1b, n4);
        n4 = (M2 * Cin) / 4;
        convert_bf16_vec4<<<(n4 + 255) / 256, 256, 0, stream>>>(x2, a2b, n4);
        transpose_w_bf16<<<(Cout * NOUT) / 256, 256, 0, stream>>>(W1, w1t, Cout);
        transpose_w_bf16<<<(Cin  * NOUT) / 256, 256, 0, stream>>>(W2, w2t, Cin);
    }

    // ---- GEMMs (TDM-staged, WMMA) ----
    gemm_tdm_bias_stats<<<dim3(M1 / 64, NOUT / 64), 128, 0, stream>>>(
        a1b, w1t, b1, y1, st1, M1, Cout);
    gemm_tdm_bias_stats<<<dim3(M2 / 64, NOUT / 64), 128, 0, stream>>>(
        a2b, w2t, b2, y2, st2, M2, Cin);

    finalize_stats<<<1, NOUT, 0, stream>>>(st1, g1, be1, sc1, sh1, 1.0f / (float)M1);
    finalize_stats<<<1, NOUT, 0, stream>>>(st2, g2, be2, sc2, sh2, 1.0f / (float)M2);

    knn3_kernel<<<B * (N1 / 256), 256, 0, stream>>>(p1, p2, N1, N2, nni, nnw);

    interp_skip_kernel<<<B * N1, NOUT, 0, stream>>>(
        y1, y2, nni, nnw, sc1, sh1, sc2, sh2, (float*)d_out);
}
